// CompressedAttention_86878598463929
// MI455X (gfx1250) — matrix operand order repaired and checked by reading, not compiled, not run on hardware
//
#include <hip/hip_runtime.h>
#include <hip/hip_bf16.h>
#include <stdint.h>

// CDNA5 (gfx1250, wave32) WMMA implementation of MLA-style compressed attention.
// - All GEMMs + attention score/PV products: v_wmma_f32_16x16x32_f16 (f32 accum)
// - Main GEMM: 16x64 tile per wave, register double-buffered K loop
// - Attention: V chunks staged LDS via global_load_async_to_lds_b128 (ASYNCcnt),
//   overlapped with the score WMMAs; s_wait_asynccnt before P.V.

typedef __attribute__((ext_vector_type(16))) _Float16 v16h;
typedef __attribute__((ext_vector_type(8)))  _Float16 v8h;
typedef __attribute__((ext_vector_type(8)))  float    v8f;

// ---------------------------------------------------------------------------
// WMMA helpers (ISA 7.12.2 layouts, wave32)
// ---------------------------------------------------------------------------

__device__ __forceinline__ v8f wmma_f16(v16h a, v16h b, v8f c) {
  // (neg_a, A, neg_b, B, c_mod, C, reuse_a, reuse_b)
  return __builtin_amdgcn_wmma_f32_16x16x32_f16(false, a, false, b, (short)0, c,
                                                false, false);
}

// A tile 16x32 f16. rowptr = &A[row(l&15)][k0]. Lane hi-group (l>=16) takes
// K 16..31; lo-group takes K 0..15.
__device__ __forceinline__ v16h load_a_tile(const _Float16* rowptr, int hi) {
  v8h lo = *(const v8h*)(rowptr + 16 * hi);
  v8h hh = *(const v8h*)(rowptr + 16 * hi + 8);
  v16h a;
#pragma unroll
  for (int i = 0; i < 8; ++i) { a[i] = lo[i]; a[i + 8] = hh[i]; }
  return a;
}

// B tile 32x16 f16: lane holds column (l&15); halves = 16 consecutive K values
// starting at 16*hi. Caller passes the fully-resolved per-lane pointer.
__device__ __forceinline__ v16h load_b_tile(const _Float16* p) {
  return *(const v16h*)p;  // 32 contiguous bytes
}

// B tile gathered with a stride (used for V chunks staged in LDS).
__device__ __forceinline__ v16h load_b_strided(const _Float16* p, int stride) {
  v16h b;
#pragma unroll
  for (int t = 0; t < 16; ++t) b[t] = p[t * stride];
  return b;
}

// ---------------------------------------------------------------------------
// Elementwise prep kernels
// ---------------------------------------------------------------------------

__global__ void cvt_f32_to_f16(const float* __restrict__ src,
                               _Float16* __restrict__ dst, int n) {
  int i = blockIdx.x * blockDim.x + threadIdx.x;
  if (i < n) dst[i] = (_Float16)src[i];
}

// Wt[n][k] = (f16) W[k][n]   (so WMMA B-tiles load contiguously along K)
__global__ void transpose_f32_to_f16(const float* __restrict__ W,
                                     _Float16* __restrict__ Wt, int K, int N) {
  int i = blockIdx.x * blockDim.x + threadIdx.x;
  if (i < K * N) {
    int k = i / N, n = i - k * N;
    Wt[(size_t)n * K + k] = (_Float16)W[i];
  }
}

// RoPE (d_R = 32, half-split convention of the reference), f32 in -> f16 out.
__global__ void rope_f32_to_f16(const float* __restrict__ raw,
                                _Float16* __restrict__ out, int rows, int T) {
  int i = blockIdx.x * blockDim.x + threadIdx.x;
  if (i >= rows * 16) return;
  int row = i >> 4, j = i & 15;
  int t = row % T;  // row = b*T + t
  float inv = __powf(10000.0f, -(float)j * (1.0f / 16.0f));
  float fr = (float)t * inv;
  float s, c;
  __sincosf(fr, &s, &c);
  float x1 = raw[row * 32 + j];
  float x2 = raw[row * 32 + 16 + j];
  out[row * 32 + j]      = (_Float16)(x1 * c - x2 * s);
  out[row * 32 + 16 + j] = (_Float16)(x2 * c + x1 * s);
}

// ---------------------------------------------------------------------------
// Main WMMA GEMM: C[M,N] = A[M,K] * Bt(N,K)^T, 16x64 tile per wave.
// A tile reused across 4 B tiles (4 WMMAs per K step); K loop double-buffered
// in registers so global_load_b128 latency overlaps the matrix pipe.
// Requires K % 32 == 0 (and K >= 32), N % 64 == 0.
// ---------------------------------------------------------------------------

__global__ __launch_bounds__(32) void gemm_wmma_f16_n64(
    const _Float16* __restrict__ A, const _Float16* __restrict__ Bt,
    float* __restrict__ Cf, _Float16* __restrict__ Ch, int M, int N, int K) {
  const int bn = blockIdx.x, bm = blockIdx.y;
  const int lane = threadIdx.x & 31;
  const int hi = lane >> 4, lr = lane & 15;

  const _Float16* arow = A + (size_t)(bm * 16 + lr) * K;
  const _Float16* b0r = Bt + (size_t)(bn * 64 + lr) * K + 16 * hi;
  const _Float16* b1r = b0r + (size_t)16 * K;
  const _Float16* b2r = b0r + (size_t)32 * K;
  const _Float16* b3r = b0r + (size_t)48 * K;

  v8f acc0 = {}, acc1 = {}, acc2 = {}, acc3 = {};

  // prologue loads (K-chunk 0)
  v16h a  = load_a_tile(arow, hi);
  v16h b0 = load_b_tile(b0r);
  v16h b1 = load_b_tile(b1r);
  v16h b2 = load_b_tile(b2r);
  v16h b3 = load_b_tile(b3r);

  for (int kk = 32; kk < K; kk += 32) {
    // issue next chunk's loads before consuming the current chunk
    v16h an  = load_a_tile(arow + kk, hi);
    v16h nb0 = load_b_tile(b0r + kk);
    v16h nb1 = load_b_tile(b1r + kk);
    v16h nb2 = load_b_tile(b2r + kk);
    v16h nb3 = load_b_tile(b3r + kk);
    __builtin_prefetch(arow + kk + 32, 0, 1);
    acc0 = wmma_f16(a, b0, acc0);
    acc1 = wmma_f16(a, b1, acc1);
    acc2 = wmma_f16(a, b2, acc2);
    acc3 = wmma_f16(a, b3, acc3);
    a = an; b0 = nb0; b1 = nb1; b2 = nb2; b3 = nb3;
  }
  acc0 = wmma_f16(a, b0, acc0);
  acc1 = wmma_f16(a, b1, acc1);
  acc2 = wmma_f16(a, b2, acc2);
  acc3 = wmma_f16(a, b3, acc3);

#pragma unroll
  for (int r = 0; r < 8; ++r) {
    size_t m = (size_t)(bm * 16 + r + 8 * hi);
    size_t n = (size_t)(bn * 64 + lr);
    if (Cf) {
      Cf[m * N + n]      = acc0[r];
      Cf[m * N + n + 16] = acc1[r];
      Cf[m * N + n + 32] = acc2[r];
      Cf[m * N + n + 48] = acc3[r];
    }
    if (Ch) {
      Ch[m * N + n]      = (_Float16)acc0[r];
      Ch[m * N + n + 16] = (_Float16)acc1[r];
      Ch[m * N + n + 32] = (_Float16)acc2[r];
      Ch[m * N + n + 48] = (_Float16)acc3[r];
    }
  }
}

// Narrow GEMM (one 16x16 tile per wave) for the N=32 rope projections.
__global__ __launch_bounds__(32) void gemm_wmma_f16_n16(
    const _Float16* __restrict__ A, const _Float16* __restrict__ Bt,
    float* __restrict__ Cf, int M, int N, int K) {
  const int bn = blockIdx.x, bm = blockIdx.y;
  const int lane = threadIdx.x & 31;
  const int hi = lane >> 4, lr = lane & 15;

  const _Float16* arow = A + (size_t)(bm * 16 + lr) * K;
  const _Float16* brow = Bt + (size_t)(bn * 16 + lr) * K + 16 * hi;

  v8f acc = {};
  for (int kk = 0; kk < K; kk += 32) {
    v16h a = load_a_tile(arow + kk, hi);
    v16h b = load_b_tile(brow + kk);
    acc = wmma_f16(a, b, acc);
  }
#pragma unroll
  for (int r = 0; r < 8; ++r) {
    size_t m = (size_t)(bm * 16 + r + 8 * hi);
    size_t n = (size_t)(bn * 16 + lr);
    Cf[m * N + n] = acc[r];
  }
}

// ---------------------------------------------------------------------------
// Flash attention with shared RoPE logits.
// One wave per (batch, head, 16-query tile). HD=64, DR=32, NH=16.
//   S = (q.kT + qr.krT) / sqrt(96), causal, online softmax, O = softmax(S).V
// V chunks are staged Global->LDS with async DMA (ASYNCcnt) so the copy runs
// underneath the score WMMAs + softmax VALU work of the same iteration.
// ---------------------------------------------------------------------------

__global__ __launch_bounds__(32) void mla_attn_kernel(
    const _Float16* __restrict__ qg, const _Float16* __restrict__ kg,
    const _Float16* __restrict__ vg, const _Float16* __restrict__ qrg,
    const _Float16* __restrict__ krg, _Float16* __restrict__ attn, int T) {
  const int QT = T >> 4;
  const int blk = blockIdx.x;
  const int qt = blk % QT;
  const int h  = (blk / QT) & 15;
  const int b  = blk / (QT * 16);

  const int lane = threadIdx.x & 31;
  const int hi = lane >> 4, lr = lane & 15;

  __shared__ _Float16 lds_p[16 * 32];   // P tile (C-layout -> A-layout shuffle)
  __shared__ _Float16 lds_v[32 * 64];   // staged V chunk (32 keys x 64 dims)

  const int q0 = qt * 16;
  const size_t rowbase = (size_t)(b * T + q0);

  // Query tiles (held in registers for the whole key loop)
  const _Float16* qrow = qg + (rowbase + lr) * 1024 + h * 64;
  v16h qa0 = load_a_tile(qrow, hi);        // head dims 0..31
  v16h qa1 = load_a_tile(qrow + 32, hi);   // head dims 32..63
  v16h qra = load_a_tile(qrg + (rowbase + lr) * 32, hi);  // rope dims 0..31

  // Per-lane LDS byte address for async V staging (ISA 10.2: LDS = addr[31:0])
  const unsigned lds_vdst = (unsigned)(uintptr_t)(lds_v + lane * 64);

  v8f o0 = {}, o1 = {}, o2 = {}, o3 = {};
  float rmax[8], rsum[8];
#pragma unroll
  for (int r = 0; r < 8; ++r) { rmax[r] = -3.0e38f; rsum[r] = 0.0f; }

  const float invs = 0.10206207261596575f;  // 1/sqrt(64+32)

  for (int kc = 0; kc < q0 + 16; kc += 32) {
    // Kick off async Global->LDS DMA of V[kc..kc+31][0..31] for this head.
    // 64 B per lane = 4x b128; tracked by ASYNCcnt, overlaps the score math.
    {
      const _Float16* vsrc = vg + ((size_t)(b * T + kc + lane)) * 1024 + h * 64;
      asm volatile(
          "global_load_async_to_lds_b128 %0, %1, off\n\t"
          "global_load_async_to_lds_b128 %0, %1, off offset:16\n\t"
          "global_load_async_to_lds_b128 %0, %1, off offset:32\n\t"
          "global_load_async_to_lds_b128 %0, %1, off offset:48"
          :
          : "v"(lds_vdst), "v"(vsrc)
          : "memory");
    }

    // Score tiles: s0 = keys kc..kc+15, s1 = keys kc+16..kc+31
    const _Float16* kb0 = kg + ((size_t)(b * T + kc + lr)) * 1024 + h * 64 + 16 * hi;
    const _Float16* kb1 = kb0 + (size_t)16 * 1024;
    v8f s0 = {}, s1 = {};
    s0 = wmma_f16(qa0, load_b_tile(kb0), s0);
    s0 = wmma_f16(qa1, load_b_tile(kb0 + 32), s0);
    s1 = wmma_f16(qa0, load_b_tile(kb1), s1);
    s1 = wmma_f16(qa1, load_b_tile(kb1 + 32), s1);
    const _Float16* krb0 = krg + ((size_t)(b * T + kc + lr)) * 32 + 16 * hi;
    s0 = wmma_f16(qra, load_b_tile(krb0), s0);
    s1 = wmma_f16(qra, load_b_tile(krb0 + 16 * 32), s1);

    // Online softmax over the 32-key chunk (rows spread over 16-lane groups).
#pragma unroll
    for (int r = 0; r < 8; ++r) {
      const int qm = q0 + r + 8 * hi;           // global query row of this slot
      float x0 = s0[r] * invs;
      float x1 = s1[r] * invs;
      if (kc + lr > qm)      x0 = -3.0e38f;     // causal mask
      if (kc + 16 + lr > qm) x1 = -3.0e38f;

      float mloc = fmaxf(x0, x1);
#pragma unroll
      for (int d = 1; d < 16; d <<= 1) mloc = fmaxf(mloc, __shfl_xor(mloc, d, 32));
      float mnew = fmaxf(rmax[r], mloc);
      float sf = __expf(rmax[r] - mnew);
      rmax[r] = mnew;

      float p0 = __expf(x0 - mnew);
      float p1 = __expf(x1 - mnew);
      float ps = p0 + p1;
#pragma unroll
      for (int d = 1; d < 16; d <<= 1) ps += __shfl_xor(ps, d, 32);
      rsum[r] = rsum[r] * sf + ps;

      o0[r] *= sf; o1[r] *= sf; o2[r] *= sf; o3[r] *= sf;

      lds_p[(r + 8 * hi) * 32 + lr]      = (_Float16)p0;
      lds_p[(r + 8 * hi) * 32 + lr + 16] = (_Float16)p1;
    }

    // V chunk must be resident in LDS before the P.V WMMAs read it.
    asm volatile("s_wait_asynccnt 0x0" ::: "memory");
    __syncthreads();

    // O += P(16x32) * V(32x64) : one A tile, four 16-col B tiles from LDS.
    v16h pa = load_a_tile(lds_p + lr * 32, hi);
    o0 = wmma_f16(pa, load_b_strided(lds_v + 16 * hi * 64 + lr,      64), o0);
    o1 = wmma_f16(pa, load_b_strided(lds_v + 16 * hi * 64 + 16 + lr, 64), o1);
    o2 = wmma_f16(pa, load_b_strided(lds_v + 16 * hi * 64 + 32 + lr, 64), o2);
    o3 = wmma_f16(pa, load_b_strided(lds_v + 16 * hi * 64 + 48 + lr, 64), o3);
    __syncthreads();
  }

#pragma unroll
  for (int r = 0; r < 8; ++r) {
    float inv = 1.0f / rsum[r];
    size_t orow = (rowbase + r + 8 * hi) * 1024 + h * 64;
    attn[orow + lr]      = (_Float16)(o0[r] * inv);
    attn[orow + 16 + lr] = (_Float16)(o1[r] * inv);
    attn[orow + 32 + lr] = (_Float16)(o2[r] * inv);
    attn[orow + 48 + lr] = (_Float16)(o3[r] * inv);
  }
}

// ---------------------------------------------------------------------------
// Host orchestration
// ---------------------------------------------------------------------------

extern "C" void kernel_launch(void* const* d_in, const int* in_sizes, int n_in,
                              void* d_out, int out_size, void* d_ws, size_t ws_size,
                              hipStream_t stream) {
  (void)in_sizes; (void)n_in; (void)out_size; (void)ws_size;

  const int B = 2, T = 2048, DM = 1024, DC = 128, DR = 32;
  const int M = B * T;  // 4096 token rows

  const float* x         = (const float*)d_in[0];
  const float* W_kv_down = (const float*)d_in[1];  // (DM, DC)
  const float* W_k_up    = (const float*)d_in[2];  // (DC, DM)
  const float* W_v_up    = (const float*)d_in[3];  // (DC, DM)
  const float* W_q       = (const float*)d_in[4];  // (DM, DM)
  const float* W_q_rope  = (const float*)d_in[5];  // (DM, DR)
  const float* W_k_rope  = (const float*)d_in[6];  // (DC, DR)
  const float* W_o       = (const float*)d_in[7];  // (DM, DM)
  float* out = (float*)d_out;                      // (B, T, DM) f32

  // Workspace carve-out (256B aligned slices)
  char* w = (char*)d_ws;
  auto take = [&](size_t bytes) {
    void* p = (void*)w;
    w += (bytes + 255) & ~(size_t)255;
    return p;
  };
  _Float16* x16     = (_Float16*)take((size_t)M * DM * 2);
  _Float16* ckv16   = (_Float16*)take((size_t)M * DC * 2);
  _Float16* q16     = (_Float16*)take((size_t)M * DM * 2);
  _Float16* k16     = (_Float16*)take((size_t)M * DM * 2);
  _Float16* v16v    = (_Float16*)take((size_t)M * DM * 2);
  _Float16* attn16  = (_Float16*)take((size_t)M * DM * 2);
  float*    qr_raw  = (float*)take((size_t)M * DR * 4);
  float*    kr_raw  = (float*)take((size_t)M * DR * 4);
  _Float16* qr16    = (_Float16*)take((size_t)M * DR * 2);
  _Float16* kr16    = (_Float16*)take((size_t)M * DR * 2);
  _Float16* Wt_kvd  = (_Float16*)take((size_t)DM * DC * 2);
  _Float16* Wt_kup  = (_Float16*)take((size_t)DC * DM * 2);
  _Float16* Wt_vup  = (_Float16*)take((size_t)DC * DM * 2);
  _Float16* Wt_q    = (_Float16*)take((size_t)DM * DM * 2);
  _Float16* Wt_qr   = (_Float16*)take((size_t)DM * DR * 2);
  _Float16* Wt_kr   = (_Float16*)take((size_t)DC * DR * 2);
  _Float16* Wt_o    = (_Float16*)take((size_t)DM * DM * 2);

  const int TB = 256;
  auto blocks = [&](int n) { return (n + TB - 1) / TB; };

  // --- precision conversion / weight transposes ---
  cvt_f32_to_f16<<<blocks(M * DM), TB, 0, stream>>>(x, x16, M * DM);
  transpose_f32_to_f16<<<blocks(DM * DC), TB, 0, stream>>>(W_kv_down, Wt_kvd, DM, DC);
  transpose_f32_to_f16<<<blocks(DC * DM), TB, 0, stream>>>(W_k_up,    Wt_kup, DC, DM);
  transpose_f32_to_f16<<<blocks(DC * DM), TB, 0, stream>>>(W_v_up,    Wt_vup, DC, DM);
  transpose_f32_to_f16<<<blocks(DM * DM), TB, 0, stream>>>(W_q,       Wt_q,   DM, DM);
  transpose_f32_to_f16<<<blocks(DM * DR), TB, 0, stream>>>(W_q_rope,  Wt_qr,  DM, DR);
  transpose_f32_to_f16<<<blocks(DC * DR), TB, 0, stream>>>(W_k_rope,  Wt_kr,  DC, DR);
  transpose_f32_to_f16<<<blocks(DM * DM), TB, 0, stream>>>(W_o,       Wt_o,   DM, DM);

  // --- projection GEMMs (16x64-per-wave WMMA) ---
  // c_kv = x @ W_kv_down
  gemm_wmma_f16_n64<<<dim3(DC / 64, M / 16), 32, 0, stream>>>(x16, Wt_kvd, nullptr, ckv16, M, DC, DM);
  // k = c_kv @ W_k_up ; v = c_kv @ W_v_up
  gemm_wmma_f16_n64<<<dim3(DM / 64, M / 16), 32, 0, stream>>>(ckv16, Wt_kup, nullptr, k16, M, DM, DC);
  gemm_wmma_f16_n64<<<dim3(DM / 64, M / 16), 32, 0, stream>>>(ckv16, Wt_vup, nullptr, v16v, M, DM, DC);
  // q = x @ W_q
  gemm_wmma_f16_n64<<<dim3(DM / 64, M / 16), 32, 0, stream>>>(x16, Wt_q, nullptr, q16, M, DM, DM);
  // rope projections (narrow N=32, f32 out for accurate rotation)
  gemm_wmma_f16_n16<<<dim3(DR / 16, M / 16), 32, 0, stream>>>(x16, Wt_qr, qr_raw, M, DR, DM);
  gemm_wmma_f16_n16<<<dim3(DR / 16, M / 16), 32, 0, stream>>>(ckv16, Wt_kr, kr_raw, M, DR, DC);

  // --- rotary embedding ---
  rope_f32_to_f16<<<blocks(M * 16), TB, 0, stream>>>(qr_raw, qr16, M, T);
  rope_f32_to_f16<<<blocks(M * 16), TB, 0, stream>>>(kr_raw, kr16, M, T);

  // --- causal flash attention with shared rope logits ---
  mla_attn_kernel<<<B * 16 * (T / 16), 32, 0, stream>>>(q16, k16, v16v, qr16, kr16, attn16, T);

  // --- output projection: out = attn @ W_o (f32 result) ---
  gemm_wmma_f16_n64<<<dim3(DM / 64, M / 16), 32, 0, stream>>>(attn16, Wt_o, out, nullptr, M, DM, DM);
}
